// PointNetSetAbstraction_61323543052297
// MI455X (gfx1250) — matrix operand order repaired and checked by reading, compiled, not run on hardware
//
#include <hip/hip_runtime.h>
#include <cstdint>
#include <cstddef>

// Problem constants (match reference)
#define B_   16
#define N_   4096
#define S_   512
#define K_   32
#define D_   6
#define R2_  0.04f            // RADIUS^2
#define EPS_ 1e-5f
#define ROWS_ ((size_t)(B_) * (S_) * (K_))   // 262144 rows into the MLP

// ---------------- CDNA5 WMMA types ----------------
typedef __attribute__((ext_vector_type(16))) __bf16 v16bf;
typedef __attribute__((ext_vector_type(8)))  float  v8f;

union Frag16 { unsigned short u[16]; v16bf v; };
union FragC  { float f[8];           v8f   v; };

// ---------------- bf16 helpers (RNE) ----------------
__device__ __forceinline__ unsigned short f2bf(float x) {
  unsigned int u = __float_as_uint(x);
  u += 0x7FFFu + ((u >> 16) & 1u);
  return (unsigned short)(u >> 16);
}
__device__ __forceinline__ float bf2f(unsigned short h) {
  return __uint_as_float(((unsigned int)h) << 16);
}

// =====================================================================
// Kernel 1: farthest point sampling. One block per batch.
// xyz layout (B,3,N). Writes new_xyz both transposed into d_out[(B,3,S)]
// and as (B*S,3) scratch for the ball-query kernel.
// =====================================================================
__global__ void __launch_bounds__(256)
fps_kernel(const float* __restrict__ xyz, float* __restrict__ out1,
           float* __restrict__ newxyz) {
  __shared__ float sx[N_], sy[N_], sz[N_];
  __shared__ float redv[256];
  __shared__ int   redi[256];

  const int b = blockIdx.x;
  const int t = threadIdx.x;
  const float* xb = xyz + (size_t)b * 3 * N_;

  for (int i = t; i < N_; i += 256) {
    sx[i] = xb[i];
    sy[i] = xb[N_ + i];
    sz[i] = xb[2 * N_ + i];
  }
  float dist[16];
#pragma unroll
  for (int j = 0; j < 16; ++j) dist[j] = 1e10f;
  __syncthreads();

  int far = 0;
  for (int it = 0; it < S_; ++it) {
    const float cx = sx[far], cy = sy[far], cz = sz[far];
    if (t == 0) {
      float* o = out1 + (size_t)b * 3 * S_;
      o[0 * S_ + it] = cx; o[1 * S_ + it] = cy; o[2 * S_ + it] = cz;
      float* nx = newxyz + ((size_t)b * S_ + it) * 3;
      nx[0] = cx; nx[1] = cy; nx[2] = cz;
    }
    float bv = -1.0f; int bi = 0;
#pragma unroll
    for (int j = 0; j < 16; ++j) {
      const int i = t + 256 * j;
      const float dx = sx[i] - cx, dy = sy[i] - cy, dz = sz[i] - cz;
      const float d = dx * dx + dy * dy + dz * dz;
      const float nd = fminf(dist[j], d);
      dist[j] = nd;
      if (nd > bv) { bv = nd; bi = i; }
    }
    redv[t] = bv; redi[t] = bi;
    __syncthreads();
    for (int off = 128; off > 0; off >>= 1) {
      if (t < off) {
        const float v2 = redv[t + off]; const int i2 = redi[t + off];
        if (v2 > redv[t] || (v2 == redv[t] && i2 < redi[t])) {
          redv[t] = v2; redi[t] = i2;
        }
      }
      __syncthreads();
    }
    far = redi[0];
    __syncthreads();
  }
}

// =====================================================================
// Kernel 2: ball query + grouping. One block per (b,s).
// Writes X0: (B*S*K) rows x 32 bf16 channels (ch0..2 = xyz-center,
// ch3..8 = point feats, ch9..31 = 0 padding for the K=32 WMMA chunk).
// =====================================================================
__global__ void __launch_bounds__(256)
ballquery_group_kernel(const float* __restrict__ xyz, const float* __restrict__ pts,
                       const float* __restrict__ newxyz,
                       unsigned short* __restrict__ X0) {
  __shared__ float d[N_];
  __shared__ float redv[256];
  __shared__ int   redi[256];
  __shared__ int   selIdx[K_];
  __shared__ float selD[K_];

  const int bs = blockIdx.x;            // b*S + s
  const int b = bs >> 9;
  const int t = threadIdx.x;
  const float* xb = xyz + (size_t)b * 3 * N_;
  const float* c3 = newxyz + (size_t)bs * 3;
  const float cx = c3[0], cy = c3[1], cz = c3[2];

  for (int i = t; i < N_; i += 256) {
    const float dx = xb[i] - cx, dy = xb[N_ + i] - cy, dz = xb[2 * N_ + i] - cz;
    d[i] = dx * dx + dy * dy + dz * dz;
  }
  __syncthreads();

  for (int k = 0; k < K_; ++k) {
    float bv = __builtin_inff(); int bi = 0;
#pragma unroll
    for (int j = 0; j < 16; ++j) {
      const int i = t + 256 * j;
      const float v = d[i];
      if (v < bv) { bv = v; bi = i; }
    }
    redv[t] = bv; redi[t] = bi;
    __syncthreads();
    for (int off = 128; off > 0; off >>= 1) {
      if (t < off) {
        const float v2 = redv[t + off]; const int i2 = redi[t + off];
        if (v2 < redv[t] || (v2 == redv[t] && i2 < redi[t])) {
          redv[t] = v2; redi[t] = i2;
        }
      }
      __syncthreads();
    }
    if (t == 0) {
      selIdx[k] = redi[0];
      selD[k]   = redv[0];
      d[redi[0]] = __builtin_inff();
    }
    __syncthreads();
  }

  const float* pb = pts + (size_t)b * D_ * N_;
  if (t < K_) {
    const int k = t;
    const int idx = (selD[k] > R2_) ? selIdx[0] : selIdx[k];
    unsigned short* xr = X0 + ((size_t)bs * K_ + k) * 32;
    xr[0] = f2bf(xb[idx] - cx);
    xr[1] = f2bf(xb[N_ + idx] - cy);
    xr[2] = f2bf(xb[2 * N_ + idx] - cz);
#pragma unroll
    for (int c = 0; c < D_; ++c) xr[3 + c] = f2bf(pb[c * N_ + idx]);
#pragma unroll
    for (int c = 3 + D_; c < 32; ++c) xr[c] = 0;
  }
}

// =====================================================================
// Kernel 3: WMMA GEMM  Hpre = act(X) * W^T + bias (+ per-channel
// sum/sumsq for batchnorm). Template params => fully unrolled fragment
// loops (registers, no scratch) and compile-time LDS sizing.
//
// LDS is staged fragment-major so each lane reads its whole 32B
// fragment with 2x ds_load_b128:
//  A: blob (rowgroup, kc): slot = m + 16*half, elem j  (K = j+8*(j>>3)+8*half)
//  B: blob (nt, kc):       slot = n + 16*half, elem j  (K = j+8*(j>>3)+8*half)
// =====================================================================
template <int CINPAD, int CINLOG, int COUT, bool APPLYBN>
__global__ void __launch_bounds__(256)
gemm_bn_kernel(const unsigned short* __restrict__ Xin,
               const float* __restrict__ W, const float* __restrict__ bias,
               const float* __restrict__ scale, const float* __restrict__ shift,
               unsigned short* __restrict__ Hout,
               float* __restrict__ gsum, float* __restrict__ gsum2) {
  constexpr int NK = CINPAD / 32;   // K-chunks of 32
  constexpr int NN = COUT / 16;     // N-tiles of 16

  __shared__ v16bf AldsV[8 * NK * 32];   // 8 row-groups
  __shared__ v16bf BldsV[NN * NK * 32];
  __shared__ float biasLds[COUT];
  __shared__ float scLds[CINPAD];
  __shared__ float shLds[CINPAD];
  __shared__ float sSum[COUT];
  __shared__ float sSum2[COUT];

  unsigned short* Alds = (unsigned short*)AldsV;
  unsigned short* Blds = (unsigned short*)BldsV;

  const int t    = threadIdx.x;
  const int lane = t & 31;
  const int wave = t >> 5;
  const int half = lane >> 4;
  const int l16  = lane & 15;
  const size_t rowBase0 = (size_t)blockIdx.x * 128;

  __builtin_prefetch(Xin + rowBase0 * CINPAD, 0, 3);   // global_prefetch_b8

  for (int i = t; i < COUT; i += 256) {
    biasLds[i] = bias[i]; sSum[i] = 0.f; sSum2[i] = 0.f;
  }
  if (APPLYBN) {
    for (int i = t; i < CINPAD; i += 256) { scLds[i] = scale[i]; shLds[i] = shift[i]; }
  }
  // Stage W swizzled: (o,k) -> blob(nt=o>>4, kc=k>>5), slot=(o&15)+16*half, elem j (table order of k&31)
  for (int i = t; i < COUT * CINPAD; i += 256) {
    const int o = i / CINPAD, k = i - o * CINPAD;
    const unsigned short val =
        (k < CINLOG) ? f2bf(W[o * CINLOG + k]) : (unsigned short)0;
    const int nt = o >> 4, n = o & 15, kc = k >> 5, L = k & 31;
    const int h = (L >> 3) & 1, j = L - 8 * h - 8 * (L >> 4);
    Blds[(((nt * NK + kc) * 32) + n + 16 * h) * 16 + j] = val;
  }
  __syncthreads();
  // Stage X swizzled (+ BN/ReLU of previous layer when APPLYBN):
  // (r,c) -> blob(rowgrp=r>>4, kc=c>>5); kin=c&31:
  //   half=(kin>>3)&1; j = kin - 8*half - 8*(kin>>4); slot = (r&15)+16*half
  for (int i = t; i < 128 * CINPAD; i += 256) {
    const int r = i / CINPAD, c = i - r * CINPAD;
    unsigned short raw = Xin[(rowBase0 + r) * CINPAD + c];
    if (APPLYBN) {
      float x = bf2f(raw);
      x = fmaxf(scLds[c] * x + shLds[c], 0.f);
      raw = f2bf(x);
    }
    const int rowgrp = r >> 4, m = r & 15, kc = c >> 5, kin = c & 31;
    const int h = (kin >> 3) & 1;
    const int j = kin - 8 * h - 8 * (kin >> 4);
    Alds[(((rowgrp * NK + kc) * 32) + m + 16 * h) * 16 + j] = raw;
  }
  __syncthreads();

  // A fragments for this wave (vector LDS loads -> ds_load_b128 pairs)
  Frag16 afr[NK];
#pragma unroll
  for (int kc = 0; kc < NK; ++kc)
    afr[kc].v = AldsV[(wave * NK + kc) * 32 + lane];

#pragma unroll
  for (int nt = 0; nt < NN; ++nt) {
    FragC c;
    const float bb = biasLds[nt * 16 + l16];
#pragma unroll
    for (int v = 0; v < 8; ++v) c.f[v] = bb;
#pragma unroll
    for (int kc = 0; kc < NK; ++kc) {
      Frag16 bfr;
      bfr.v = BldsV[(nt * NK + kc) * 32 + lane];
      c.v = __builtin_amdgcn_wmma_f32_16x16x32_bf16(
          false, afr[kc].v, false, bfr.v, (short)0, c.v, false, false);
    }
    // store bf16 pre-BN activations + per-channel stats
    float s = 0.f, s2 = 0.f;
#pragma unroll
    for (int v = 0; v < 8; ++v) {
      const size_t r = rowBase0 + (size_t)wave * 16 + v + 8 * half;
      Hout[r * COUT + nt * 16 + l16] = f2bf(c.f[v]);
      s += c.f[v]; s2 += c.f[v] * c.f[v];
    }
    atomicAdd(&sSum[nt * 16 + l16], s);     // ds_add_f32
    atomicAdd(&sSum2[nt * 16 + l16], s2);
  }
  __syncthreads();
  for (int i = t; i < COUT; i += 256) {
    atomicAdd(&gsum[i], sSum[i]);
    atomicAdd(&gsum2[i], sSum2[i]);
  }
}

// =====================================================================
// Kernel 4: finalize batchnorm -> per-channel scale/shift
// =====================================================================
__global__ void bn_finalize_kernel(const float* __restrict__ gsum,
                                   const float* __restrict__ gsum2,
                                   const float* __restrict__ g,
                                   const float* __restrict__ be, int cout,
                                   float cnt, float* __restrict__ scale,
                                   float* __restrict__ shift) {
  const int c = threadIdx.x;
  if (c >= cout) return;
  const float mean = gsum[c] / cnt;
  const float var  = gsum2[c] / cnt - mean * mean;
  const float sc   = g[c] * rsqrtf(var + EPS_);
  scale[c] = sc;
  shift[c] = be[c] - mean * sc;
}

// =====================================================================
// Kernel 5: BN + ReLU + max over K, transposed store into d_out (B,128,S)
// =====================================================================
__global__ void __launch_bounds__(128)
bn_relu_maxpool_kernel(const unsigned short* __restrict__ H2,
                       const float* __restrict__ scale,
                       const float* __restrict__ shift,
                       float* __restrict__ out2) {
  const int bs = blockIdx.x;     // b*S + s
  const int o  = threadIdx.x;    // channel 0..127
  const float sc = scale[o], sh = shift[o];
  const size_t base = (size_t)bs * K_ * 128 + o;
  float m = 0.0f;                // ReLU outputs are >= 0
#pragma unroll 4
  for (int k = 0; k < K_; ++k) {
    float x = sc * bf2f(H2[base + (size_t)k * 128]) + sh;
    x = fmaxf(x, 0.f);
    m = fmaxf(m, x);
  }
  const int b = bs >> 9, s = bs & (S_ - 1);
  out2[((size_t)b * 128 + o) * S_ + s] = m;
}

// =====================================================================
// Kernel 6: zero the stats accumulators (ws is poisoned, not zeroed)
// =====================================================================
__global__ void zero_kernel(float* __restrict__ p, int n) {
  const int i = blockIdx.x * 256 + threadIdx.x;
  if (i < n) p[i] = 0.f;
}

// =====================================================================
extern "C" void kernel_launch(void* const* d_in, const int* in_sizes, int n_in,
                              void* d_out, int out_size, void* d_ws, size_t ws_size,
                              hipStream_t stream) {
  (void)in_sizes; (void)n_in; (void)out_size; (void)ws_size;
  const float* xyz = (const float*)d_in[0];
  const float* pts = (const float*)d_in[1];
  const float* W0 = (const float*)d_in[2];  const float* b0 = (const float*)d_in[3];
  const float* g0 = (const float*)d_in[4];  const float* be0 = (const float*)d_in[5];
  const float* W1 = (const float*)d_in[6];  const float* b1 = (const float*)d_in[7];
  const float* g1 = (const float*)d_in[8];  const float* be1 = (const float*)d_in[9];
  const float* W2 = (const float*)d_in[10]; const float* b2 = (const float*)d_in[11];
  const float* g2 = (const float*)d_in[12]; const float* be2 = (const float*)d_in[13];

  char* ws = (char*)d_ws;
  size_t off = 0;
  auto carve = [&](size_t bytes) -> char* {
    char* p = ws + off;
    off = (off + bytes + 255) & ~(size_t)255;
    return p;
  };
  float*          newxyz = (float*)carve((size_t)B_ * S_ * 3 * sizeof(float));
  unsigned short* X0 = (unsigned short*)carve(ROWS_ * 32 * sizeof(unsigned short));
  unsigned short* H0 = (unsigned short*)carve(ROWS_ * 64 * sizeof(unsigned short));
  unsigned short* H1 = (unsigned short*)carve(ROWS_ * 64 * sizeof(unsigned short));
  unsigned short* H2 = (unsigned short*)carve(ROWS_ * 128 * sizeof(unsigned short));
  float*          stats = (float*)carve(6 * 128 * sizeof(float));  // sum/sumsq x3
  float*          scsh  = (float*)carve(6 * 128 * sizeof(float));  // scale/shift x3

  float* out1 = (float*)d_out;              // (B,3,S)
  float* out2 = out1 + (size_t)B_ * 3 * S_; // (B,128,S)

  const int nblk = (int)(ROWS_ / 128);      // 2048
  const float cnt = (float)ROWS_;

  fps_kernel<<<B_, 256, 0, stream>>>(xyz, out1, newxyz);
  ballquery_group_kernel<<<B_ * S_, 256, 0, stream>>>(xyz, pts, newxyz, X0);
  zero_kernel<<<3, 256, 0, stream>>>(stats, 6 * 128);

  // Layer 0: 9 (padded 32) -> 64
  gemm_bn_kernel<32, 9, 64, false><<<nblk, 256, 0, stream>>>(
      X0, W0, b0, nullptr, nullptr, H0, stats + 0, stats + 128);
  bn_finalize_kernel<<<1, 64, 0, stream>>>(stats + 0, stats + 128, g0, be0, 64,
                                           cnt, scsh + 0, scsh + 128);
  // Layer 1: 64 -> 64 (applies BN0+ReLU on input)
  gemm_bn_kernel<64, 64, 64, true><<<nblk, 256, 0, stream>>>(
      H0, W1, b1, scsh + 0, scsh + 128, H1, stats + 256, stats + 384);
  bn_finalize_kernel<<<1, 64, 0, stream>>>(stats + 256, stats + 384, g1, be1, 64,
                                           cnt, scsh + 256, scsh + 384);
  // Layer 2: 64 -> 128 (applies BN1+ReLU on input)
  gemm_bn_kernel<64, 64, 128, true><<<nblk, 256, 0, stream>>>(
      H1, W2, b2, scsh + 256, scsh + 384, H2, stats + 512, stats + 640);
  bn_finalize_kernel<<<1, 128, 0, stream>>>(stats + 512, stats + 640, g2, be2, 128,
                                            cnt, scsh + 512, scsh + 640);
  // BN2 + ReLU + max over K, transposed store
  bn_relu_maxpool_kernel<<<B_ * S_, 128, 0, stream>>>(H2, scsh + 512, scsh + 640, out2);
}
